// KinematicSfxnNetwork_15436112462614
// MI455X (gfx1250) — matrix operand order repaired and checked by reading, not compiled, not run on hardware
//
#include <hip/hip_runtime.h>
#include <hip/hip_bf16.h>

#define P_CONST 8
#define A_CONST 65536
#define N_CONST (P_CONST * A_CONST)   // 524288
#define N1_CONST (N_CONST + 1)        // 524289
#define BLK 64
#define NB_CONST ((N1_CONST + BLK - 1) / BLK)  // 8193

typedef _Float16 v16h __attribute__((ext_vector_type(16)));
typedef float    v8f  __attribute__((ext_vector_type(8)));

// ---------------------------------------------------------------------------
// Pass 1: per-block serial window scan. One thread handles 64 nodes.
// Emits per-node (alpha, R rel-to-entering-window) and per-block transfer.
// R is an affine 3x4 stored row-major: [r00 r01 r02 t0; r10 ...; r20 ... t2].
// ---------------------------------------------------------------------------
__global__ void k_pass1(const float* __restrict__ masked,
                        const float* __restrict__ base,
                        const int*   __restrict__ parents,
                        float* __restrict__ Rn, int* __restrict__ alphaArr,
                        int* __restrict__ Ta, float* __restrict__ TR)
{
    int b = blockIdx.x * blockDim.x + threadIdx.x;
    if (b >= NB_CONST) return;
    int s = b * BLK;

    float w0[12], w1[12], w2[12], w3[12];
    int a0, a1, a2, a3;
#pragma unroll
    for (int e = 0; e < 12; e++) { w0[e] = w1[e] = w2[e] = w3[e] = 0.f; }
    w0[0] = w0[5] = w0[10] = 1.f;  a0 = 0;
    w1[0] = w1[5] = w1[10] = 1.f;  a1 = 1;
    w2[0] = w2[5] = w2[10] = 1.f;  a2 = 2;
    w3[0] = w3[5] = w3[10] = 1.f;  a3 = 3;

    int iend = s + BLK; if (iend > N1_CONST) iend = N1_CONST;

    for (int i = s; i < iend; i++) {
        float Ri[12]; int ai;
        if (i == 0) {
#pragma unroll
            for (int e = 0; e < 12; e++) Ri[e] = 0.f;
            Ri[0] = Ri[5] = Ri[10] = 1.f;
            ai = 0;
        } else {
            float phi = masked[i];
            float th  = base[i * 4 + 1];
            float dd  = base[i * 4 + 2];
            float sp = __sinf(phi), cp = __cosf(phi);
            float st = __sinf(th),  ct = __cosf(th);
            float m[12];
            m[0] = cp * ct; m[1] = -sp; m[2]  = cp * st; m[3]  = dd * m[0];
            m[4] = sp * ct; m[5] =  cp; m[6]  = sp * st; m[7]  = dd * m[4];
            m[8] = -st;     m[9] = 0.f; m[10] = ct;      m[11] = dd * m[8];

            int p = parents[i];
            int slot = p & 3;
            float Xp[12];
            switch (slot) {
                case 0: { for (int e = 0; e < 12; e++) Xp[e] = w0[e]; ai = a0; } break;
                case 1: { for (int e = 0; e < 12; e++) Xp[e] = w1[e]; ai = a1; } break;
                case 2: { for (int e = 0; e < 12; e++) Xp[e] = w2[e]; ai = a2; } break;
                default:{ for (int e = 0; e < 12; e++) Xp[e] = w3[e]; ai = a3; } break;
            }
#pragma unroll
            for (int r = 0; r < 3; r++) {
                float x0 = Xp[r*4+0], x1 = Xp[r*4+1], x2 = Xp[r*4+2], x3 = Xp[r*4+3];
                Ri[r*4+0] = x0*m[0] + x1*m[4] + x2*m[8];
                Ri[r*4+1] = x0*m[1] + x1*m[5] + x2*m[9];
                Ri[r*4+2] = x0*m[2] + x1*m[6] + x2*m[10];
                Ri[r*4+3] = x0*m[3] + x1*m[7] + x2*m[11] + x3;
            }
        }
        alphaArr[i] = ai;
#pragma unroll
        for (int e = 0; e < 12; e++) Rn[(size_t)i * 12 + e] = Ri[e];

        switch (i & 3) {
            case 0: { for (int e = 0; e < 12; e++) w0[e] = Ri[e]; a0 = ai; } break;
            case 1: { for (int e = 0; e < 12; e++) w1[e] = Ri[e]; a1 = ai; } break;
            case 2: { for (int e = 0; e < 12; e++) w2[e] = Ri[e]; a2 = ai; } break;
            default:{ for (int e = 0; e < 12; e++) w3[e] = Ri[e]; a3 = ai; } break;
        }
    }
    Ta[b*4+0] = a0; Ta[b*4+1] = a1; Ta[b*4+2] = a2; Ta[b*4+3] = a3;
#pragma unroll
    for (int e = 0; e < 12; e++) {
        TR[((size_t)b*4 + 0)*12 + e] = w0[e];
        TR[((size_t)b*4 + 1)*12 + e] = w1[e];
        TR[((size_t)b*4 + 2)*12 + e] = w2[e];
        TR[((size_t)b*4 + 3)*12 + e] = w3[e];
    }
}

// ---------------------------------------------------------------------------
// Pass 2: Hillis-Steele inclusive scan over block transfers.
// Compose X (earlier) then Y (later): a[j]=X.a[Y.a[j]], R[j]=X.R[Y.a[j]]*Y.R[j]
// ---------------------------------------------------------------------------
__global__ void k_scan(const int* __restrict__ TaS, const float* __restrict__ TRS,
                       int* __restrict__ TaD, float* __restrict__ TRD, int off)
{
    int b = blockIdx.x * blockDim.x + threadIdx.x;
    if (b >= NB_CONST) return;
    if (b < off) {
#pragma unroll
        for (int j = 0; j < 4; j++) {
            TaD[b*4+j] = TaS[b*4+j];
            for (int e = 0; e < 12; e++)
                TRD[((size_t)b*4+j)*12+e] = TRS[((size_t)b*4+j)*12+e];
        }
        return;
    }
    int bx = b - off;
#pragma unroll
    for (int j = 0; j < 4; j++) {
        int ya = TaS[b*4+j];
        const float* Y = &TRS[((size_t)b*4  + j )*12];
        const float* X = &TRS[((size_t)bx*4 + ya)*12];
        TaD[b*4+j] = TaS[bx*4 + ya];
        float O[12];
#pragma unroll
        for (int r = 0; r < 3; r++) {
            float x0 = X[r*4+0], x1 = X[r*4+1], x2 = X[r*4+2], x3 = X[r*4+3];
            O[r*4+0] = x0*Y[0] + x1*Y[4] + x2*Y[8];
            O[r*4+1] = x0*Y[1] + x1*Y[5] + x2*Y[9];
            O[r*4+2] = x0*Y[2] + x1*Y[6] + x2*Y[10];
            O[r*4+3] = x0*Y[3] + x1*Y[7] + x2*Y[11] + x3;
        }
        for (int e = 0; e < 12; e++) TRD[((size_t)b*4+j)*12+e] = O[e];
    }
}

// ---------------------------------------------------------------------------
__global__ void k_copy(const float* __restrict__ src, float* __restrict__ dst, int n)
{
    for (int i = blockIdx.x * blockDim.x + threadIdx.x; i < n;
         i += gridDim.x * blockDim.x)
        dst[i] = src[i];
}

// ---------------------------------------------------------------------------
// Pass 3: per-node fixup + scatter. coord_i = E.rot * R_i.t + E.t,
// where E = S_{b-1}.R[alpha_i] (identity for block 0).
// ---------------------------------------------------------------------------
__global__ void k_pass3(const float* __restrict__ Rn, const int* __restrict__ alphaArr,
                        const float* __restrict__ TRfin,
                        const int* __restrict__ kin_id, float* __restrict__ coords)
{
    int i = blockIdx.x * blockDim.x + threadIdx.x;
    if (i < 1 || i >= N1_CONST) return;
    int b = i >> 6;
    float tx = Rn[(size_t)i*12 + 3];
    float ty = Rn[(size_t)i*12 + 7];
    float tz = Rn[(size_t)i*12 + 11];
    float cx, cy, cz;
    if (b == 0) {
        cx = tx; cy = ty; cz = tz;
    } else {
        int a = alphaArr[i];
        const float* E = &TRfin[(((size_t)(b-1))*4 + a)*12];
        cx = E[0]*tx + E[1]*ty + E[2]*tz  + E[3];
        cy = E[4]*tx + E[5]*ty + E[6]*tz  + E[7];
        cz = E[8]*tx + E[9]*ty + E[10]*tz + E[11];
    }
    int kid = kin_id[i - 1];
    coords[(size_t)kid*3 + 0] = cx;
    coords[(size_t)kid*3 + 1] = cy;
    coords[(size_t)kid*3 + 2] = cz;
}

// ---------------------------------------------------------------------------
// Energy via WMMA + async-to-LDS staging.
// Each wave owns a 129-point (1548 B) coordinate window per chunk, staged
// into LDS with 97 disjoint global_load_async_to_lds_b128 transfers
// (ASYNCcnt-tracked), then consumed lane-locally. diag(A x B) with A rows =
// 8 packed bond displacements (K=4q..4q+3 hold dx,dy,dz,0 of bond q) and
// B cols = w-scaled displacements: one v_wmma_f32_16x16x32_f16 reduces 128
// bonds, f32 accumulation chained through C. 512 chunks / 256 waves per pose
// -> exactly 2 fully unrolled iterations, double-buffered in LDS.
// ---------------------------------------------------------------------------
__global__ void __launch_bounds__(256)
k_energy(const float* __restrict__ coords, const float* __restrict__ bw,
         float* __restrict__ partial)
{
    const int NBonds = A_CONST - 1;           // 65535
    int p    = blockIdx.x >> 5;   // 32 workgroups per pose
    int g32  = blockIdx.x & 31;
    int tid  = threadIdx.x;
    int wave = tid >> 5;
    int lane = tid & 31;
    int gw   = g32 * 8 + wave;    // 0..255 waves per pose
    int grp  = lane & 15;         // group (row/col) 0..15
    bool hi  = lane >= 31 ? true : (lane >= 16);

    const float* C = coords + (size_t)p * A_CONST * 3;
    const float* W = bw     + (size_t)p * NBonds;

    __shared__ __align__(16) float stage[8][2][392]; // 129 pts * 3 = 387 fl, padded
    __shared__ float sd[256];

    v8f acc = {0.f, 0.f, 0.f, 0.f, 0.f, 0.f, 0.f, 0.f};

#pragma unroll
    for (int it = 0; it < 2; it++) {
        int c = gw + it * 256;                 // chunk index, always < 512
        const char* gsrc = (const char*)(C + (size_t)c * 128 * 3);
        char* lb = (char*)&stage[wave][it][0];

        // Stage 1548B window as 97 x 16B async global->LDS transfers.
#pragma unroll
        for (int r = 0; r < 4; r++) {
            int idx = r * 32 + lane;
            if (idx < 97) {
                unsigned    ldsOff = (unsigned)(uintptr_t)(lb + idx * 16);
                const void* ga     = (const void*)(gsrc + idx * 16);
                asm volatile("global_load_async_to_lds_b128 %0, %1, off"
                             :: "v"(ldsOff), "v"(ga) : "memory");
            }
        }
        if (it == 0) // warm the second chunk's weights while waiting
            __builtin_prefetch(W + (size_t)(gw + 256) * 128, 0, 0);
        asm volatile("s_wait_asynccnt 0x0" ::: "memory");

        int jbase = c * 128 + grp * 8;         // first bond of this lane's group
        const float* S = &stage[wave][it][grp * 8 * 3];

        float px[9], py[9], pz[9];
#pragma unroll
        for (int q = 0; q < 9; q++) {
            px[q] = S[q*3 + 0];
            py[q] = S[q*3 + 1];
            pz[q] = S[q*3 + 2];
        }
        float dx[8], dy[8], dz[8], wv[8];
#pragma unroll
        for (int q = 0; q < 8; q++) {
            int j = jbase + q;
            bool v = (j < NBonds);
            dx[q] = v ? (px[q+1] - px[q]) : 0.f;
            dy[q] = v ? (py[q+1] - py[q]) : 0.f;
            dz[q] = v ? (pz[q+1] - pz[q]) : 0.f;
            wv[q] = v ? W[j] : 0.f;
        }
        // A-matrix lanes: lo lanes carry bonds {0,1,2,3}, hi lanes {4,5,6,7}
        int qa0 = hi ? 4 : 0, qa1 = hi ? 5 : 1, qa2 = hi ? 6 : 2, qa3 = hi ? 7 : 3;
        // B-matrix lanes: lo lanes carry bonds {0,1,2,3}, hi lanes {4,5,6,7}
        int qb = hi ? 4 : 0;
        v16h aA, aB;
        aA[0]  = (_Float16)dx[qa0]; aA[1]  = (_Float16)dy[qa0];
        aA[2]  = (_Float16)dz[qa0]; aA[3]  = (_Float16)0.f;
        aA[4]  = (_Float16)dx[qa1]; aA[5]  = (_Float16)dy[qa1];
        aA[6]  = (_Float16)dz[qa1]; aA[7]  = (_Float16)0.f;
        aA[8]  = (_Float16)dx[qa2]; aA[9]  = (_Float16)dy[qa2];
        aA[10] = (_Float16)dz[qa2]; aA[11] = (_Float16)0.f;
        aA[12] = (_Float16)dx[qa3]; aA[13] = (_Float16)dy[qa3];
        aA[14] = (_Float16)dz[qa3]; aA[15] = (_Float16)0.f;
#pragma unroll
        for (int q = 0; q < 4; q++) {
            int qq = qb + q;
            aB[4*q+0] = (_Float16)(wv[qq] * dx[qq]);
            aB[4*q+1] = (_Float16)(wv[qq] * dy[qq]);
            aB[4*q+2] = (_Float16)(wv[qq] * dz[qq]);
            aB[4*q+3] = (_Float16)0.f;
        }
        acc = __builtin_amdgcn_wmma_f32_16x16x32_f16(
            false, aA, false, aB, (short)0, acc, false, false);
    }

    // diagonal of the 16x16 f32 D: lanes 0-7 hold D[r][r] in acc[r],
    // lanes 24-31 hold D[8+r][8+r] in acc[r].
    float val = 0.f;
    if (lane < 8)        val = acc[lane];
    else if (lane >= 24) val = acc[lane - 24];

    sd[tid] = val;
    __syncthreads();
    if (lane == 0) {
        float s = 0.f;
        for (int k = 0; k < 32; k++) s += sd[wave * 32 + k];
        sd[wave * 32] = s;
    }
    __syncthreads();
    if (tid == 0) {
        float s = 0.f;
        for (int wv2 = 0; wv2 < 8; wv2++) s += sd[wv2 * 32];
        partial[p * 32 + g32] = s;
    }
}

__global__ void k_final(const float* __restrict__ partial, float* __restrict__ out)
{
    int p = threadIdx.x;
    if (p < P_CONST) {
        float s = 0.f;
        for (int g = 0; g < 32; g++) s += partial[p * 32 + g];
        out[p] = s;
    }
}

// ---------------------------------------------------------------------------
extern "C" void kernel_launch(void* const* d_in, const int* in_sizes, int n_in,
                              void* d_out, int out_size, void* d_ws, size_t ws_size,
                              hipStream_t stream)
{
    const float* masked  = (const float*)d_in[0];  // N1
    const float* base    = (const float*)d_in[1];  // N1 x 4
    const float* pose    = (const float*)d_in[2];  // P x A x 3
    const float* bw      = (const float*)d_in[3];  // P x (A-1)
    const int*   parents = (const int*)d_in[4];    // N1
    const int*   kin     = (const int*)d_in[5];    // N
    float* out = (float*)d_out;

    char* w = (char*)d_ws;
    auto align256 = [](size_t x) { return (x + 255) & ~(size_t)255; };
    float* Rn     = (float*)w; w += align256((size_t)N1_CONST * 12 * 4);
    int*   alphaA = (int*)w;   w += align256((size_t)N1_CONST * 4);
    int*   Ta0    = (int*)w;   w += align256((size_t)NB_CONST * 4 * 4);
    float* TR0    = (float*)w; w += align256((size_t)NB_CONST * 48 * 4);
    int*   Ta1    = (int*)w;   w += align256((size_t)NB_CONST * 4 * 4);
    float* TR1    = (float*)w; w += align256((size_t)NB_CONST * 48 * 4);
    float* coords = (float*)w; w += align256((size_t)N_CONST * 3 * 4);
    float* partial= (float*)w; w += align256((size_t)P_CONST * 32 * 4);

    // Pass 1: per-block window scan
    k_pass1<<<(NB_CONST + 63) / 64, 64, 0, stream>>>(
        masked, base, parents, Rn, alphaA, Ta0, TR0);

    // Pass 2: inclusive scan over 8193 block transfers (14 ping-pong steps)
    int* TaB[2] = {Ta0, Ta1};
    float* TRB[2] = {TR0, TR1};
    int cur = 0;
    for (int off = 1; off < NB_CONST; off <<= 1) {
        k_scan<<<(NB_CONST + 255) / 256, 256, 0, stream>>>(
            TaB[cur], TRB[cur], TaB[1 ^ cur], TRB[1 ^ cur], off);
        cur ^= 1;
    }

    // Pass 3: init coords from pose_coords, then fixup + scatter
    k_copy<<<512, 256, 0, stream>>>(pose, coords, N_CONST * 3);
    k_pass3<<<(N1_CONST + 255) / 256, 256, 0, stream>>>(
        Rn, alphaA, TRB[cur], kin, coords);

    // Energy: async-staged WMMA contraction + deterministic reduction
    k_energy<<<P_CONST * 32, 256, 0, stream>>>(coords, bw, partial);
    k_final<<<1, 64, 0, stream>>>(partial, out);
}
